// BiLSTM_CRF_NER_86577950752825
// MI455X (gfx1250) — compile-verified, matrix operand order checked by reading, not run on hardware
//
#include <hip/hip_runtime.h>
#include <hip/hip_bf16.h>

// BiLSTM-CRF NER forward (NLL) for gfx1250.
//   1. cast weights f32->bf16 (L2-resident, ~18MB bf16)
//   2. embedding gather -> Xbf [T*B, 256] bf16
//   3. WMMA GEMM (async-to-LDS double buffered): Gx = X @ Wih^T + (bih+bhh)
//   4. lstm_scan: 16 WGs per direction; Whh slice resident in LDS (128KB,
//      async-preloaded once); h ping-pongs through L2 with one agent-scope
//      atomic barrier per timestep; recurrent GEMM via WMMA bf16.
//   5. combine fwd/bwd halves, WMMA emission GEMM (NT padded to 16)
//   6. single-block CRF forward DP -> scalar NLL
//
// Input order: 0 tokens, 1 bio_gold, 2 emb, 3..6 l0f(wih,whh,bih,bhh),
// 7..10 l0b, 11..14 l1f, 15..18 l1b, 19 emw, 20 emb_b, 21 start, 22 end, 23 trans

typedef __attribute__((ext_vector_type(16))) __bf16 v16bf;
typedef __attribute__((ext_vector_type(8)))  float  v8f;

union FragBF { v16bf v; unsigned u[8]; };

__device__ __forceinline__ unsigned short f2bf(float f) {
  unsigned u = __float_as_uint(f);
  u += 0x7FFFu + ((u >> 16) & 1u);          // round-to-nearest-even
  return (unsigned short)(u >> 16);
}
__device__ __forceinline__ float bf2f(unsigned short h) {
  return __uint_as_float(((unsigned)h) << 16);
}
// K index held by fragment element pair (2v,2v+1): {0,2,4,6,16,18,20,22} + 8*half
__device__ __forceinline__ int kofs(int v, int half) {
  return ((v < 4) ? (2 * v) : (8 + 2 * v)) + 8 * half;
}
// LDS byte offset of a __shared__ location (low 32 bits of the generic addr
// are the workgroup-relative LDS offset on gfx1250 aperture addressing).
__device__ __forceinline__ unsigned lds_off(const void* p) {
  return (unsigned)(unsigned long long)p;
}
// ---- gfx1250 async global->LDS copies (ASYNCcnt-tracked, bypass VGPRs) ----
__device__ __forceinline__ void async_b64(unsigned lds, const void* g) {
  asm volatile("global_load_async_to_lds_b64 %0, %1, off" :: "v"(lds), "v"(g) : "memory");
}
__device__ __forceinline__ void async_b128(unsigned lds, const void* g) {
  asm volatile("global_load_async_to_lds_b128 %0, %1, off" :: "v"(lds), "v"(g) : "memory");
}
template <int N>
__device__ __forceinline__ void wait_async() {
  asm volatile("s_wait_asynccnt %0" :: "i"(N) : "memory");
}

// ---------------------------------------------------------------- utilities
__global__ void zero_u32_k(unsigned* __restrict__ p, int n) {
  int i = blockIdx.x * 256 + threadIdx.x;
  if (i < n) p[i] = 0u;
}
__global__ void cast_f32_bf16_k(const float* __restrict__ s,
                                unsigned short* __restrict__ d, int n) {
  int i = blockIdx.x * 256 + threadIdx.x;
  if (i < n) d[i] = f2bf(s[i]);
}
__global__ void embed_gather_k(const int* __restrict__ tokens,
                               const float* __restrict__ emb,
                               unsigned short* __restrict__ X) {
  int bx = blockIdx.x;            // row = t*32 + b (time-major)
  int t = bx >> 5, b = bx & 31;
  int tok = tokens[b * 256 + t];
  X[(size_t)bx * 256 + threadIdx.x] = f2bf(emb[(size_t)tok * 256 + threadIdx.x]);
}
__global__ void combine_avg_k(const unsigned short* __restrict__ H1,
                              unsigned short* __restrict__ O) {
  int bx = blockIdx.x;
  int t = bx >> 5, b = bx & 31;
  for (int j = threadIdx.x; j < 512; j += 256) {
    float f = bf2f(H1[(size_t)bx * 1024 + j]);
    float w = bf2f(H1[(size_t)bx * 1024 + 512 + j]);
    O[(size_t)(b * 256 + t) * 512 + j] = f2bf(0.5f * (f + w));
  }
}

// -------------------------------------------------- big input-projection GEMM
// C[M,N] = A[M,K]bf16 @ W[N,K]bf16^T + bias0 + bias1  (block tile 32x256,
// double-buffered async global->LDS staging; padded LDS rows vs bank conflicts)
__global__ __launch_bounds__(256) void gemm_bf16_nt(
    const unsigned short* __restrict__ A, const unsigned short* __restrict__ W,
    const float* __restrict__ bias0, const float* __restrict__ bias1,
    float* __restrict__ C, int M, int N, int K) {
  __shared__ __attribute__((aligned(16))) unsigned short lds_a[2][32 * 40];
  __shared__ __attribute__((aligned(16))) unsigned short lds_w[2][256 * 40];
  const int tid = threadIdx.x;
  const int wv = tid >> 5, lane = tid & 31;
  const int half = lane >> 4, rl = lane & 15;
  const int bm0 = blockIdx.x * 32;
  const int bn0 = blockIdx.y * 256;

  const unsigned short* gA = &A[(size_t)(bm0 + (tid >> 3)) * K + (tid & 7) * 4];
  const unsigned short* gW = &W[(size_t)(bn0 + tid) * K];
  unsigned la[2], lw[2];
  la[0] = lds_off(&lds_a[0][(tid >> 3) * 40 + (tid & 7) * 4]);
  la[1] = lds_off(&lds_a[1][(tid >> 3) * 40 + (tid & 7) * 4]);
  lw[0] = lds_off(&lds_w[0][tid * 40]);
  lw[1] = lds_off(&lds_w[1][tid * 40]);

  const int nch = K >> 5;
  // prologue: chunk 0 into buffer 0 (1x b64 + 4x b128 per thread)
  async_b64(la[0], gA);
#pragma unroll
  for (int i = 0; i < 4; ++i) async_b128(lw[0] + i * 16, (const char*)gW + i * 16);

  v8f acc00 = {}, acc01 = {}, acc10 = {}, acc11 = {};
  for (int c = 0; c < nch; ++c) {
    const int cur = c & 1;
    if (c + 1 < nch) {
      async_b64(la[1 - cur], gA + (c + 1) * 32);
#pragma unroll
      for (int i = 0; i < 4; ++i)
        async_b128(lw[1 - cur] + i * 16, (const char*)(gW + (c + 1) * 32) + i * 16);
      wait_async<5>();
    } else {
      wait_async<0>();
    }
    __syncthreads();
    FragBF a0, a1, b0, b1;
#pragma unroll
    for (int v = 0; v < 8; ++v) {
      int k = kofs(v, half);
      a0.u[v] = *(const unsigned*)&lds_a[cur][rl * 40 + k];
      a1.u[v] = *(const unsigned*)&lds_a[cur][(16 + rl) * 40 + k];
      b0.u[v] = *(const unsigned*)&lds_w[cur][(wv * 32 + rl) * 40 + k];
      b1.u[v] = *(const unsigned*)&lds_w[cur][(wv * 32 + 16 + rl) * 40 + k];
    }
    acc00 = __builtin_amdgcn_wmma_f32_16x16x32_bf16(false, a0.v, false, b0.v, (short)0, acc00, false, false);
    acc01 = __builtin_amdgcn_wmma_f32_16x16x32_bf16(false, a0.v, false, b1.v, (short)0, acc01, false, false);
    acc10 = __builtin_amdgcn_wmma_f32_16x16x32_bf16(false, a1.v, false, b0.v, (short)0, acc10, false, false);
    acc11 = __builtin_amdgcn_wmma_f32_16x16x32_bf16(false, a1.v, false, b1.v, (short)0, acc11, false, false);
    __syncthreads();
  }
#pragma unroll
  for (int v = 0; v < 8; ++v) {
    int m = v + 8 * half;
    int c0 = bn0 + wv * 32 + rl, c1 = c0 + 16;
    float bv0 = bias0[c0] + bias1[c0];
    float bv1 = bias0[c1] + bias1[c1];
    C[(size_t)(bm0 + m) * N + c0] = acc00[v] + bv0;
    C[(size_t)(bm0 + m) * N + c1] = acc01[v] + bv1;
    C[(size_t)(bm0 + 16 + m) * N + c0] = acc10[v] + bv0;
    C[(size_t)(bm0 + 16 + m) * N + c1] = acc11[v] + bv1;
  }
}

// ------------------------------------------------------------ recurrent scan
// grid = 32 blocks: dir = bx/16, each of 16 WGs owns 32 hidden units (128 gate
// rows).  The WG's Whh slice (128x512 bf16 = 128KB) is async-preloaded into
// LDS once and stays resident for all 256 timesteps (LDS rows padded to 520).
#define SCAN_NWG 16
__global__ __launch_bounds__(256) void lstm_scan_k(
    const float* __restrict__ GxF, const float* __restrict__ GxB,
    const unsigned short* __restrict__ WhhF, const unsigned short* __restrict__ WhhB,
    unsigned short* __restrict__ Hcur, unsigned short* __restrict__ Hout,
    int* __restrict__ cnt) {
  __shared__ __attribute__((aligned(16))) unsigned short lds_whh[128 * 520];
  __shared__ __attribute__((aligned(16))) unsigned short lds_h[2][32 * 40];
  __shared__ float gates[32 * 128];

  const int tid = threadIdx.x;
  const int wv = tid >> 5, lane = tid & 31;
  const int half = lane >> 4, rl = lane & 15;
  const int dir = blockIdx.x >> 4;
  const int hbase = (blockIdx.x & 15) * 32;
  const float* Gx = dir ? GxB : GxF;
  const unsigned short* Whh = dir ? WhhB : WhhF;
  int* mycnt = cnt + dir;
  unsigned short* hc = Hcur + (size_t)dir * (2 * 32 * 512);

  {   // one-shot Whh slice preload: thread t -> row n = t/2, 256-elem segment
    int n = tid >> 1, seg = (tid & 1) * 256;
    int wr = (n >> 5) * 512 + hbase + (n & 31);   // gate col n -> weight row
    const char* g = (const char*)&Whh[(size_t)wr * 512 + seg];
    unsigned l = lds_off(&lds_whh[n * 520 + seg]);
#pragma unroll
    for (int i = 0; i < 32; ++i) async_b128(l + i * 16, g + i * 16);
    wait_async<0>();
  }
  __syncthreads();

  unsigned lh[2];
  lh[0] = lds_off(&lds_h[0][(tid >> 3) * 40 + (tid & 7) * 4]);
  lh[1] = lds_off(&lds_h[1][(tid >> 3) * 40 + (tid & 7) * 4]);
  const int r_st = tid >> 3, c_st = (tid & 7) * 4;

  float cst[4] = {0.f, 0.f, 0.f, 0.f};   // c-state, 4 (b,j) items per thread

  for (int t = 0; t < 256; ++t) {
    const int t_eff = dir ? (255 - t) : t;
    const unsigned short* hread = hc + (t & 1) * (32 * 512);
    v8f acc0 = {}, acc1 = {};
    async_b64(lh[0], &hread[r_st * 512 + c_st]);        // chunk 0
    for (int kk = 0; kk < 16; ++kk) {
      const int cur = kk & 1;
      if (kk + 1 < 16) {
        async_b64(lh[1 - cur], &hread[r_st * 512 + (kk + 1) * 32 + c_st]);
        wait_async<1>();
      } else {
        wait_async<0>();
      }
      __syncthreads();
      FragBF a0, a1, bf;
#pragma unroll
      for (int v = 0; v < 8; ++v) {
        int k = kofs(v, half);
        a0.u[v] = *(const unsigned*)&lds_h[cur][rl * 40 + k];
        a1.u[v] = *(const unsigned*)&lds_h[cur][(16 + rl) * 40 + k];
        bf.u[v] = *(const unsigned*)&lds_whh[(wv * 16 + rl) * 520 + kk * 32 + k];
      }
      acc0 = __builtin_amdgcn_wmma_f32_16x16x32_bf16(false, a0.v, false, bf.v, (short)0, acc0, false, false);
      acc1 = __builtin_amdgcn_wmma_f32_16x16x32_bf16(false, a1.v, false, bf.v, (short)0, acc1, false, false);
      __syncthreads();
    }
#pragma unroll
    for (int v = 0; v < 8; ++v) {              // spill C tiles: rows=batch
      int m = v + 8 * half;
      gates[m * 128 + wv * 16 + rl] = acc0[v];
      gates[(16 + m) * 128 + wv * 16 + rl] = acc1[v];
    }
    __syncthreads();
    unsigned short* hwrite = hc + ((t + 1) & 1) * (32 * 512);
#pragma unroll
    for (int i = 0; i < 4; ++i) {              // LSTM cell (i,f,g,o order)
      int q = tid + 256 * i;
      int b = q >> 5, j = q & 31;
      const float* gx = &Gx[(size_t)(t_eff * 32 + b) * 2048 + hbase + j];
      float gi = gates[b * 128 + j]        + gx[0];
      float gf = gates[b * 128 + 32 + j]   + gx[512];
      float gg = gates[b * 128 + 64 + j]   + gx[1024];
      float go = gates[b * 128 + 96 + j]   + gx[1536];
      float si = 1.f / (1.f + __expf(-gi));
      float sf = 1.f / (1.f + __expf(-gf));
      float so = 1.f / (1.f + __expf(-go));
      float c = sf * cst[i] + si * tanhf(gg);
      cst[i] = c;
      unsigned short hb = f2bf(so * tanhf(c));
      hwrite[b * 512 + hbase + j] = hb;
      Hout[(size_t)(t_eff * 32 + b) * 1024 + dir * 512 + hbase + j] = hb;
    }
    // direction-wide barrier (16 WGs), one per timestep: release by all
    // threads (threadfence), arrival/acquire spin by lane 0, then an
    // acq-rel fence by ALL threads so every lane's next-step reads of the
    // other WGPs' h-writes see fresh lines (not stale WGP$ from t-1).
    __threadfence();
    __syncthreads();
    if (tid == 0) {
      __hip_atomic_fetch_add(mycnt, 1, __ATOMIC_RELEASE, __HIP_MEMORY_SCOPE_AGENT);
      int target = SCAN_NWG * (t + 1);
      while (__hip_atomic_load(mycnt, __ATOMIC_ACQUIRE, __HIP_MEMORY_SCOPE_AGENT) < target)
        __builtin_amdgcn_s_sleep(1);
    }
    __syncthreads();
    __threadfence();   // per-thread acquire: invalidate stale h lines
  }
}

// ----------------------------------------------------------- emission GEMM
// C[8192,16] = A[8192,512]bf16 @ Wp[16,512]^T + bias(pad 9->16)
__global__ __launch_bounds__(256) void emis_gemm_k(
    const unsigned short* __restrict__ A, const unsigned short* __restrict__ Wp,
    const float* __restrict__ bias, float* __restrict__ C) {
  __shared__ __attribute__((aligned(16))) unsigned short lds_w[16 * 520];
  __shared__ float lbias[16];
  const int tid = threadIdx.x;
  const int wv = tid >> 5, lane = tid & 31;
  const int half = lane >> 4, rl = lane & 15;
  {
    int r = tid >> 4, cs = (tid & 15) * 32;
    const char* g = (const char*)&Wp[r * 512 + cs];
    unsigned l = lds_off(&lds_w[r * 520 + cs]);
#pragma unroll
    for (int i = 0; i < 4; ++i) async_b128(l + i * 16, g + i * 16);
    wait_async<0>();
  }
  if (tid < 16) lbias[tid] = (tid < 9) ? bias[tid] : 0.f;
  __syncthreads();
  const int m0 = blockIdx.x * 128 + wv * 16;
  v8f acc = {};
  for (int kk = 0; kk < 512; kk += 32) {
    FragBF a, b;
#pragma unroll
    for (int v = 0; v < 8; ++v) {
      int k = kofs(v, half);
      a.u[v] = *(const unsigned*)&A[(size_t)(m0 + rl) * 512 + kk + k];
      b.u[v] = *(const unsigned*)&lds_w[rl * 520 + kk + k];
    }
    acc = __builtin_amdgcn_wmma_f32_16x16x32_bf16(false, a.v, false, b.v, (short)0, acc, false, false);
  }
#pragma unroll
  for (int v = 0; v < 8; ++v) {
    int m = v + 8 * half;
    C[(size_t)(m0 + m) * 16 + rl] = acc[v] + lbias[rl];
  }
}

// ------------------------------------------------------------------- CRF NLL
__global__ __launch_bounds__(288) void crf_nll_k(
    const float* __restrict__ E, const int* __restrict__ tokens,
    const int* __restrict__ tags, const float* __restrict__ startv,
    const float* __restrict__ endv, const float* __restrict__ trans,
    float* __restrict__ out) {
  __shared__ float sc[32][9];
  __shared__ float ltr[81], lst[9], len[9];
  __shared__ float den[32], num[32];
  const int tid = threadIdx.x;
  if (tid < 81) ltr[tid] = trans[tid];
  if (tid < 9) { lst[tid] = startv[tid]; len[tid] = endv[tid]; }
  __syncthreads();
  const int b = tid / 9, j = tid % 9;
  sc[b][j] = lst[j] + E[(size_t)(b * 256) * 16 + j];
  __syncthreads();
  for (int t = 1; t < 256; ++t) {
    bool mt = tokens[b * 256 + t] != 0;
    float mx = -1e30f;
#pragma unroll
    for (int i = 0; i < 9; ++i) mx = fmaxf(mx, sc[b][i] + ltr[i * 9 + j]);
    float s = 0.f;
#pragma unroll
    for (int i = 0; i < 9; ++i) s += __expf(sc[b][i] + ltr[i * 9 + j] - mx);
    float nxt = mx + __logf(s) + E[(size_t)(b * 256 + t) * 16 + j];
    __syncthreads();
    if (mt) sc[b][j] = nxt;
    __syncthreads();
  }
  if (j == 0) {
    float mx = -1e30f;
    for (int i = 0; i < 9; ++i) mx = fmaxf(mx, sc[b][i] + len[i]);
    float s = 0.f;
    for (int i = 0; i < 9; ++i) s += __expf(sc[b][i] + len[i] - mx);
    den[b] = mx + __logf(s);
    int prev = tags[b * 256];
    float nm = lst[prev] + E[(size_t)(b * 256) * 16 + prev];
    for (int t = 1; t < 256; ++t) {
      int tg = tags[b * 256 + t];
      if (tokens[b * 256 + t] != 0) {
        nm += ltr[prev * 9 + tg] + E[(size_t)(b * 256 + t) * 16 + tg];
        prev = tg;
      }
    }
    num[b] = nm + len[prev];
  }
  __syncthreads();
  if (tid == 0) {
    float a = 0.f;
    for (int bb = 0; bb < 32; ++bb) a += den[bb] - num[bb];
    out[0] = a / 32.f;
  }
}

// ---------------------------------------------------------------- workspace
constexpr size_t OF_HCUR = 0,                       SZ_HCUR = 2ull * 2 * 32 * 512 * 2;
constexpr size_t OF_CNT  = OF_HCUR + SZ_HCUR,       SZ_CNT  = 256;
constexpr size_t OF_EMW  = OF_CNT + SZ_CNT,         SZ_EMW  = 16ull * 512 * 2;
constexpr size_t OF_XBF  = OF_EMW + SZ_EMW,         SZ_XBF  = 8192ull * 256 * 2;
constexpr size_t OF_W0F  = OF_XBF + SZ_XBF,         SZ_W0   = 2048ull * 256 * 2;
constexpr size_t OF_W0B  = OF_W0F + SZ_W0;
constexpr size_t OF_WH0F = OF_W0B + SZ_W0,          SZ_WH   = 2048ull * 512 * 2;
constexpr size_t OF_WH0B = OF_WH0F + SZ_WH;
constexpr size_t OF_WH1F = OF_WH0B + SZ_WH;
constexpr size_t OF_WH1B = OF_WH1F + SZ_WH;
constexpr size_t OF_W1F  = OF_WH1B + SZ_WH,         SZ_W1   = 2048ull * 1024 * 2;
constexpr size_t OF_W1B  = OF_W1F + SZ_W1;
constexpr size_t OF_GXF  = OF_W1B + SZ_W1,          SZ_GX   = 8192ull * 2048 * 4;
constexpr size_t OF_GXB  = OF_GXF + SZ_GX;
constexpr size_t OF_H0   = OF_GXB + SZ_GX,          SZ_H    = 8192ull * 1024 * 2;
constexpr size_t OF_H1   = OF_H0 + SZ_H;
constexpr size_t OF_OBF  = OF_H1 + SZ_H,            SZ_OBF  = 8192ull * 512 * 2;
constexpr size_t OF_EMIS = OF_OBF + SZ_OBF;

extern "C" void kernel_launch(void* const* d_in, const int* in_sizes, int n_in,
                              void* d_out, int out_size, void* d_ws, size_t ws_size,
                              hipStream_t stream) {
  const int*   tokens = (const int*)d_in[0];
  const int*   tags   = (const int*)d_in[1];
  const float* emb    = (const float*)d_in[2];
  const float* w0f_ih = (const float*)d_in[3],  *w0f_hh = (const float*)d_in[4];
  const float* b0f_ih = (const float*)d_in[5],  *b0f_hh = (const float*)d_in[6];
  const float* w0b_ih = (const float*)d_in[7],  *w0b_hh = (const float*)d_in[8];
  const float* b0b_ih = (const float*)d_in[9],  *b0b_hh = (const float*)d_in[10];
  const float* w1f_ih = (const float*)d_in[11], *w1f_hh = (const float*)d_in[12];
  const float* b1f_ih = (const float*)d_in[13], *b1f_hh = (const float*)d_in[14];
  const float* w1b_ih = (const float*)d_in[15], *w1b_hh = (const float*)d_in[16];
  const float* b1b_ih = (const float*)d_in[17], *b1b_hh = (const float*)d_in[18];
  const float* emw    = (const float*)d_in[19];
  const float* emb_b  = (const float*)d_in[20];
  const float* c_sta  = (const float*)d_in[21];
  const float* c_end  = (const float*)d_in[22];
  const float* c_trn  = (const float*)d_in[23];

  char* ws = (char*)d_ws;
  unsigned short* Hcur = (unsigned short*)(ws + OF_HCUR);
  int*            cnts = (int*)(ws + OF_CNT);
  unsigned short* EmwP = (unsigned short*)(ws + OF_EMW);
  unsigned short* Xbf  = (unsigned short*)(ws + OF_XBF);
  unsigned short* W0F  = (unsigned short*)(ws + OF_W0F);
  unsigned short* W0B  = (unsigned short*)(ws + OF_W0B);
  unsigned short* WH0F = (unsigned short*)(ws + OF_WH0F);
  unsigned short* WH0B = (unsigned short*)(ws + OF_WH0B);
  unsigned short* WH1F = (unsigned short*)(ws + OF_WH1F);
  unsigned short* WH1B = (unsigned short*)(ws + OF_WH1B);
  unsigned short* W1F  = (unsigned short*)(ws + OF_W1F);
  unsigned short* W1B  = (unsigned short*)(ws + OF_W1B);
  float*          GxF  = (float*)(ws + OF_GXF);
  float*          GxB  = (float*)(ws + OF_GXB);
  unsigned short* H0   = (unsigned short*)(ws + OF_H0);
  unsigned short* H1   = (unsigned short*)(ws + OF_H1);
  unsigned short* Obf  = (unsigned short*)(ws + OF_OBF);
  float*          Emis = (float*)(ws + OF_EMIS);

  {   // zero Hcur + counters + EmwPad
    int nz = (int)((SZ_HCUR + SZ_CNT + SZ_EMW) / 4);
    zero_u32_k<<<(nz + 255) / 256, 256, 0, stream>>>((unsigned*)ws, nz);
  }
  cast_f32_bf16_k<<<(2048 * 256 + 255) / 256, 256, 0, stream>>>(w0f_ih, W0F, 2048 * 256);
  cast_f32_bf16_k<<<(2048 * 256 + 255) / 256, 256, 0, stream>>>(w0b_ih, W0B, 2048 * 256);
  cast_f32_bf16_k<<<(2048 * 512 + 255) / 256, 256, 0, stream>>>(w0f_hh, WH0F, 2048 * 512);
  cast_f32_bf16_k<<<(2048 * 512 + 255) / 256, 256, 0, stream>>>(w0b_hh, WH0B, 2048 * 512);
  cast_f32_bf16_k<<<(2048 * 512 + 255) / 256, 256, 0, stream>>>(w1f_hh, WH1F, 2048 * 512);
  cast_f32_bf16_k<<<(2048 * 512 + 255) / 256, 256, 0, stream>>>(w1b_hh, WH1B, 2048 * 512);
  cast_f32_bf16_k<<<(2048 * 1024 + 255) / 256, 256, 0, stream>>>(w1f_ih, W1F, 2048 * 1024);
  cast_f32_bf16_k<<<(2048 * 1024 + 255) / 256, 256, 0, stream>>>(w1b_ih, W1B, 2048 * 1024);
  cast_f32_bf16_k<<<(9 * 512 + 255) / 256, 256, 0, stream>>>(emw, EmwP, 9 * 512);

  embed_gather_k<<<8192, 256, 0, stream>>>(tokens, emb, Xbf);

  // layer 0 input projections + scan
  gemm_bf16_nt<<<dim3(8192 / 32, 2048 / 256), 256, 0, stream>>>(Xbf, W0F, b0f_ih, b0f_hh, GxF, 8192, 2048, 256);
  gemm_bf16_nt<<<dim3(8192 / 32, 2048 / 256), 256, 0, stream>>>(Xbf, W0B, b0b_ih, b0b_hh, GxB, 8192, 2048, 256);
  lstm_scan_k<<<32, 256, 0, stream>>>(GxF, GxB, WH0F, WH0B, Hcur, H0, cnts);

  // layer 1 input projections (K=1024 concat input) + scan
  gemm_bf16_nt<<<dim3(8192 / 32, 2048 / 256), 256, 0, stream>>>(H0, W1F, b1f_ih, b1f_hh, GxF, 8192, 2048, 1024);
  gemm_bf16_nt<<<dim3(8192 / 32, 2048 / 256), 256, 0, stream>>>(H0, W1B, b1b_ih, b1b_hh, GxB, 8192, 2048, 1024);
  {   // re-zero Hcur + counters for layer 1
    int nz = (int)((SZ_HCUR + SZ_CNT) / 4);
    zero_u32_k<<<(nz + 255) / 256, 256, 0, stream>>>((unsigned*)ws, nz);
  }
  lstm_scan_k<<<32, 256, 0, stream>>>(GxF, GxB, WH1F, WH1B, Hcur, H1, cnts);

  combine_avg_k<<<8192, 256, 0, stream>>>(H1, Obf);
  emis_gemm_k<<<64, 256, 0, stream>>>(Obf, EmwP, emb_b, Emis);
  crf_nll_k<<<1, 288, 0, stream>>>(Emis, tokens, tags, c_sta, c_end, c_trn, (float*)d_out);
}
